// Merge_MixtralSparseMoeBlock_9637906612735
// MI455X (gfx1250) — matrix operand order repaired and checked by reading, not compiled, not run on hardware
//
#include <hip/hip_runtime.h>
#include <cstdint>
#include <cstddef>

typedef __bf16 bf16;
typedef __attribute__((ext_vector_type(16))) __bf16 v16bf;
typedef __attribute__((ext_vector_type(8)))  __bf16 v8bf;
typedef __attribute__((ext_vector_type(4)))  __bf16 v4bf;
typedef __attribute__((ext_vector_type(8)))  float  v8f;
typedef __attribute__((ext_vector_type(4)))  unsigned int v4u;
typedef __attribute__((ext_vector_type(8)))  int v8i;
typedef __attribute__((ext_vector_type(4)))  int v4i;

// Problem dims
constexpr int Bc = 4, Sc = 2048, Tc = Bc * Sc;    // 8192 tokens
constexpr int Hc = 1024, Fc = 3584, Ec = 8, Rc = 79;
constexpr int TILE = 128;
constexpr int MAXS = Tc * 2 + Ec * TILE;          // 17408 padded expert slots
constexpr int SLOT_TILES = MAXS / TILE;           // 136
constexpr int LDK = 40;                           // 32 + 8 bf16 pad, keeps 16B align

// ---------------- helpers ----------------
__device__ inline v8f zero_v8f() {
  v8f v;
#pragma unroll
  for (int i = 0; i < 8; ++i) v[i] = 0.0f;
  return v;
}
__device__ inline v8bf zero_v8bf() {
  v8bf v;
#pragma unroll
  for (int i = 0; i < 8; ++i) v[i] = (bf16)0.0f;
  return v;
}
__device__ inline unsigned lds_off(const void* p) {
  // flat shared address: [63:32]=aperture, [31:0]=LDS byte offset
  return (unsigned)(uintptr_t)p;
}

// Issue one TDM 2-D tile load: tile_rows x 32 bf16, row stride = row_stride_elems,
// LDS destination padded 16B per 64B row -> 40-element row stride (matches LDK).
__device__ inline void tdm_issue_b(unsigned lds_byte_off, const bf16* gptr,
                                   int tile_rows, int row_stride_elems) {
  const unsigned long long ga = (unsigned long long)(uintptr_t)gptr;
  v4u g0;
  g0[0] = 1u;                                                // count=1, user desc
  g0[1] = lds_byte_off;                                      // lds_addr
  g0[2] = (unsigned)(ga & 0xFFFFFFFFu);                      // global_addr[31:0]
  g0[3] = (unsigned)((ga >> 32) & 0x1FFFFFFu) | (2u << 30);  // ga[56:32] | type=2
  v8i g1;
  // data_size=1 (2B), pad_enable=1, pad_interval=3 (16 DWORDs), pad_amount=3 (4 DWORDs)
  g1[0] = (int)((1u << 16) | (1u << 20) | (3u << 22) | (3u << 25));
  g1[1] = 0;                            // atomic barrier addr 0; tensor_dim0 lo16 (=0, td0=1<<30)
  g1[2] = 0x4000;                       // tensor_dim0 hi16 ; tensor_dim1 lo16 (td1=1<<30)
  g1[3] = 0x4000 | (32 << 16);          // tensor_dim1 hi16 ; tile_dim0 = 32 elems
  g1[4] = tile_rows;                    // tile_dim1 ; tile_dim2 = 0 (2-D)
  g1[5] = row_stride_elems;             // tensor_dim0_stride lo32
  g1[6] = 0;                            // stride hi / dim1_stride lo
  g1[7] = 0;
  const v4i z4 = {0, 0, 0, 0};
  const v8i z8 = {0, 0, 0, 0, 0, 0, 0, 0};
  // amdgpu-toolchain (clang-23) 6-arg form
  __builtin_amdgcn_tensor_load_to_lds(g0, g1, z4, z4, z8, 0);
}

// A fragment (16x32 bf16): lanes 0-15 rows M=0..15 with K {0..7,16..23};
// lanes 16-31 same rows, K {8..15,24..31}.  LDS tile is [M][K] stride LDK.
__device__ inline v16bf frag_a(const bf16* p, int lane) {
  const int row = lane & 15;
  const int c0 = (lane & 16) ? 8 : 0;
  union { v16bf v; v8bf h[2]; } u;
  u.h[0] = *(const v8bf*)(p + row * LDK + c0);
  u.h[1] = *(const v8bf*)(p + row * LDK + c0 + 16);
  return u.v;
}
// B fragment (32x16 bf16, stored as weight rows [N][K] stride LDK):
// lanes 0-15 col N=lane, K {0..7,16..23}; lanes 16-31 K {8..15,24..31}.
__device__ inline v16bf frag_b(const bf16* p, int lane) {
  const int col = lane & 15;
  const int k0 = (lane & 16) ? 8 : 0;
  union { v16bf v; v8bf h[2]; } u;
  u.h[0] = *(const v8bf*)(p + col * LDK + k0);
  u.h[1] = *(const v8bf*)(p + col * LDK + k0 + 16);
  return u.v;
}

// ---------------- setup kernels ----------------
__global__ void k_init(int* __restrict__ token_of, float* __restrict__ w_of,
                       int* __restrict__ counts, int* __restrict__ cursor) {
  const int i = blockIdx.x * 256 + threadIdx.x;
  if (i < MAXS) { token_of[i] = -1; w_of[i] = 0.0f; }
  if (i < Ec) { counts[i] = 0; cursor[i] = 0; }
}

__global__ void k_cvt(const float* __restrict__ x, bf16* __restrict__ xb) {
  const size_t i = (size_t)blockIdx.x * 256 + threadIdx.x;   // over T*H/4
  if (i >= (size_t)Tc * Hc / 4) return;
  const float4 v = ((const float4*)x)[i];
  v4bf o;
  o[0] = (bf16)v.x; o[1] = (bf16)v.y; o[2] = (bf16)v.z; o[3] = (bf16)v.w;
  ((v4bf*)xb)[i] = o;
}

// dst[e][m][n] = W[e][m][n] + sum_r U[e][m][r]*V[e][r][n]  (cast to bf16)
__global__ void k_prep(const float* __restrict__ W, const float* __restrict__ U,
                       const float* __restrict__ V, bf16* __restrict__ dst,
                       int M, int N) {
  const int e = blockIdx.y;
  const size_t mn = (size_t)blockIdx.x * 256 + threadIdx.x;
  if (mn >= (size_t)M * N) return;
  const int m = (int)(mn / N);
  const int n = (int)(mn % N);
  const float* Wp = W + (size_t)e * M * N;
  const float* Up = U + (size_t)e * M * Rc + (size_t)m * Rc;
  const float* Vp = V + (size_t)e * Rc * N;
  float acc = Wp[mn];
  for (int r = 0; r < Rc; ++r) acc += Up[r] * Vp[(size_t)r * N + n];
  dst[(size_t)e * M * N + mn] = (bf16)acc;
}

// ---------------- routing ----------------
__global__ __launch_bounds__(256) void k_router(
    const float* __restrict__ x, const float* __restrict__ gw,
    float* __restrict__ logits, int* __restrict__ sel,
    float* __restrict__ wtop, int* __restrict__ counts) {
  const int lane = threadIdx.x & 31;
  const int t = (blockIdx.x * 256 + threadIdx.x) >> 5;   // one wave per token
  if (t >= Tc) return;
  float acc[Ec];
#pragma unroll
  for (int e = 0; e < Ec; ++e) acc[e] = 0.0f;
  const float* xr = x + (size_t)t * Hc;
  for (int i = lane; i < Hc; i += 32) {
    const float xv = xr[i];
#pragma unroll
    for (int e = 0; e < Ec; ++e) acc[e] += xv * gw[e * Hc + i];
  }
#pragma unroll
  for (int e = 0; e < Ec; ++e) {
#pragma unroll
    for (int off = 16; off > 0; off >>= 1)
      acc[e] += __shfl_xor(acc[e], off, 32);
  }
  if (lane == 0) {
    float mx = acc[0];
#pragma unroll
    for (int e = 1; e < Ec; ++e) mx = fmaxf(mx, acc[e]);
    float p[Ec]; float s = 0.0f;
#pragma unroll
    for (int e = 0; e < Ec; ++e) { p[e] = __expf(acc[e] - mx); s += p[e]; }
    const float inv = 1.0f / s;
#pragma unroll
    for (int e = 0; e < Ec; ++e) {
      p[e] *= inv;
      logits[(size_t)t * Ec + e] = acc[e];
    }
    int i0 = 0;
#pragma unroll
    for (int e = 1; e < Ec; ++e) if (p[e] > p[i0]) i0 = e;
    int i1 = (i0 == 0) ? 1 : 0;
#pragma unroll
    for (int e = 0; e < Ec; ++e) if (e != i0 && p[e] > p[i1]) i1 = e;
    const float ssum = p[i0] + p[i1];
    sel[2 * t] = i0; sel[2 * t + 1] = i1;
    wtop[2 * t] = p[i0] / ssum; wtop[2 * t + 1] = p[i1] / ssum;
    atomicAdd(&counts[i0], 1); atomicAdd(&counts[i1], 1);
  }
}

__global__ void k_bases(const int* __restrict__ counts, int* __restrict__ base) {
  if (threadIdx.x == 0 && blockIdx.x == 0) {
    int acc = 0;
    for (int e = 0; e < Ec; ++e) {
      base[e] = acc;
      acc += (counts[e] + TILE - 1) & ~(TILE - 1);   // 128-align each expert region
    }
    base[Ec] = acc;
  }
}

__global__ void k_assign(const int* __restrict__ sel, const float* __restrict__ wtop,
                         const int* __restrict__ base, int* __restrict__ cursor,
                         int* __restrict__ token_of, float* __restrict__ w_of,
                         int* __restrict__ slot_of) {
  const int t = blockIdx.x * 256 + threadIdx.x;
  if (t >= Tc) return;
#pragma unroll
  for (int k = 0; k < 2; ++k) {
    const int e = sel[2 * t + k];
    const int pos = atomicAdd(&cursor[e], 1);
    const int s = base[e] + pos;
    token_of[s] = t;
    w_of[s] = wtop[2 * t + k];
    slot_of[2 * t + k] = s;
  }
}

// ---------------- pass 1: gate/up GEMM + SiLU, h = silu(x@W1e.T)*(x@W3e.T)*w ----------------
__global__ __launch_bounds__(256) void k_gemm1(
    const bf16* __restrict__ xb, const bf16* __restrict__ W1e,
    const bf16* __restrict__ W3e, const int* __restrict__ token_of,
    const float* __restrict__ w_of, const int* __restrict__ base,
    bf16* __restrict__ hbuf) {
  __shared__ bf16 lsA[TILE * LDK];
  __shared__ bf16 lsB1[2][64 * LDK];        // TDM double buffers
  __shared__ bf16 lsB3[2][64 * LDK];

  const int slot0 = blockIdx.x * TILE;
  if (token_of[slot0] < 0) return;          // fully-padded tile (uniform)
  int e = 0;
#pragma unroll
  for (int i = 1; i < Ec; ++i) if (slot0 >= base[i]) e = i;

  const int fbase = blockIdx.y * 64;
  const int tid = threadIdx.x;
  const int lane = tid & 31;
  const int wv = tid >> 5;
  const int wm = wv & 3;                    // 4 m-waves * 32 rows
  const int wn = wv >> 2;                   // 2 n-waves * 32 cols
  const int lr = tid >> 2;                  // loader row 0..63
  const int lc = (tid & 3) * 8;             // loader col {0,8,16,24}

  const bf16* W1p = W1e + ((size_t)e * Fc + fbase) * (size_t)Hc;
  const bf16* W3p = W3e + ((size_t)e * Fc + fbase) * (size_t)Hc;
  const int tok0 = token_of[slot0 + lr];
  const int tok1 = token_of[slot0 + lr + 64];

  v8f accg[2][2], accu[2][2];
#pragma unroll
  for (int mi = 0; mi < 2; ++mi)
#pragma unroll
    for (int ni = 0; ni < 2; ++ni) { accg[mi][ni] = zero_v8f(); accu[mi][ni] = zero_v8f(); }

  constexpr int KSTEPS = Hc / 32;
  // Prologue: wave 0 kicks off TDM for the first B tiles.
  if (wv == 0) {
    tdm_issue_b(lds_off(&lsB1[0][0]), W1p, 64, Hc);
    tdm_issue_b(lds_off(&lsB3[0][0]), W3p, 64, Hc);
  }

  for (int it = 0; it < KSTEPS; ++it) {
    const int k0 = it * 32;
    const int p = it & 1;

    // A tile (gathered token rows) via normal vector loads.
    v8bf a0 = zero_v8bf(), a1 = zero_v8bf();
    if (tok0 >= 0) {
      a0 = *(const v8bf*)(xb + (size_t)tok0 * Hc + k0 + lc);
      if (k0 + 32 < Hc) __builtin_prefetch(xb + (size_t)tok0 * Hc + k0 + 32 + lc, 0, 3);
    }
    if (tok1 >= 0) a1 = *(const v8bf*)(xb + (size_t)tok1 * Hc + k0 + lc);

    // Wave 0: issue next B tiles into the other buffer, then wait for current pair.
    if (wv == 0) {
      if (it + 1 < KSTEPS) {
        tdm_issue_b(lds_off(&lsB1[p ^ 1][0]), W1p + (k0 + 32), 64, Hc);
        tdm_issue_b(lds_off(&lsB3[p ^ 1][0]), W3p + (k0 + 32), 64, Hc);
        __builtin_amdgcn_s_wait_tensorcnt(2);   // current pair complete
      } else {
        __builtin_amdgcn_s_wait_tensorcnt(0);
      }
    }

    *(v8bf*)(lsA + lr * LDK + lc) = a0;
    *(v8bf*)(lsA + (lr + 64) * LDK + lc) = a1;
    __syncthreads();                            // A stored + B tiles resident

    v16bf af[2], b1f[2], b3f[2];
#pragma unroll
    for (int mi = 0; mi < 2; ++mi) af[mi] = frag_a(lsA + (wm * 32 + mi * 16) * LDK, lane);
#pragma unroll
    for (int ni = 0; ni < 2; ++ni) {
      b1f[ni] = frag_b(&lsB1[p][0] + (wn * 32 + ni * 16) * LDK, lane);
      b3f[ni] = frag_b(&lsB3[p][0] + (wn * 32 + ni * 16) * LDK, lane);
    }
#pragma unroll
    for (int mi = 0; mi < 2; ++mi)
#pragma unroll
      for (int ni = 0; ni < 2; ++ni) {
        accg[mi][ni] = __builtin_amdgcn_wmma_f32_16x16x32_bf16(
            false, af[mi], false, b1f[ni], (short)0, accg[mi][ni], false, false);
        accu[mi][ni] = __builtin_amdgcn_wmma_f32_16x16x32_bf16(
            false, af[mi], false, b3f[ni], (short)0, accu[mi][ni], false, false);
      }
    __syncthreads();                            // done reading buf p before it is reloaded
  }

  // epilogue: h = silu(gate)*up * routing_weight (weight folded here, so pass2 is a plain GEMM)
  const int rsub = (lane >> 4) << 3;        // 0 or 8
  const int ncol = lane & 15;
#pragma unroll
  for (int mi = 0; mi < 2; ++mi) {
#pragma unroll
    for (int v = 0; v < 8; ++v) {
      const int m = wm * 32 + mi * 16 + rsub + v;
      const int slot = slot0 + m;
      const float wt = w_of[slot];
#pragma unroll
      for (int ni = 0; ni < 2; ++ni) {
        const float g = accg[mi][ni][v];
        const float u = accu[mi][ni][v];
        const float hv = (g / (1.0f + __expf(-g))) * u * wt;
        hbuf[(size_t)slot * Fc + fbase + wn * 32 + ni * 16 + ncol] = (bf16)hv;
      }
    }
  }
}

// ---------------- pass 2: y = h @ W2e.T ----------------
__global__ __launch_bounds__(256) void k_gemm2(
    const bf16* __restrict__ hbuf, const bf16* __restrict__ W2e,
    const int* __restrict__ token_of, const int* __restrict__ base,
    float* __restrict__ y) {
  __shared__ bf16 lsA[TILE * LDK];
  __shared__ bf16 lsB[2][TILE * LDK];       // TDM double buffers

  const int slot0 = blockIdx.x * TILE;
  if (token_of[slot0] < 0) return;
  int e = 0;
#pragma unroll
  for (int i = 1; i < Ec; ++i) if (slot0 >= base[i]) e = i;

  const int nbase = blockIdx.y * TILE;      // over H
  const int tid = threadIdx.x;
  const int lane = tid & 31;
  const int wv = tid >> 5;
  const int wm = wv & 3;                    // 4 m-waves * 32 rows
  const int wn = wv >> 2;                   // 2 n-waves * 64 cols
  const int lr = tid >> 2;
  const int lc = (tid & 3) * 8;

  const bf16* Ap = hbuf + (size_t)slot0 * Fc;
  const bf16* Bp = W2e + ((size_t)e * Hc + nbase) * (size_t)Fc;

  v8f acc[2][4];
#pragma unroll
  for (int mi = 0; mi < 2; ++mi)
#pragma unroll
    for (int ni = 0; ni < 4; ++ni) acc[mi][ni] = zero_v8f();

  constexpr int KSTEPS = Fc / 32;
  if (wv == 0) tdm_issue_b(lds_off(&lsB[0][0]), Bp, TILE, Fc);

  for (int it = 0; it < KSTEPS; ++it) {
    const int k0 = it * 32;
    const int p = it & 1;

    const v8bf a0 = *(const v8bf*)(Ap + (size_t)lr * Fc + k0 + lc);
    const v8bf a1 = *(const v8bf*)(Ap + (size_t)(lr + 64) * Fc + k0 + lc);
    if (k0 + 32 < Fc) {
      __builtin_prefetch(Ap + (size_t)lr * Fc + k0 + 32 + lc, 0, 3);
      __builtin_prefetch(Ap + (size_t)(lr + 64) * Fc + k0 + 32 + lc, 0, 3);
    }

    if (wv == 0) {
      if (it + 1 < KSTEPS) {
        tdm_issue_b(lds_off(&lsB[p ^ 1][0]), Bp + (k0 + 32), TILE, Fc);
        __builtin_amdgcn_s_wait_tensorcnt(1);
      } else {
        __builtin_amdgcn_s_wait_tensorcnt(0);
      }
    }

    *(v8bf*)(lsA + lr * LDK + lc) = a0;
    *(v8bf*)(lsA + (lr + 64) * LDK + lc) = a1;
    __syncthreads();

    v16bf af[2], bf_[4];
#pragma unroll
    for (int mi = 0; mi < 2; ++mi) af[mi] = frag_a(lsA + (wm * 32 + mi * 16) * LDK, lane);
#pragma unroll
    for (int ni = 0; ni < 4; ++ni) bf_[ni] = frag_b(&lsB[p][0] + (wn * 64 + ni * 16) * LDK, lane);
#pragma unroll
    for (int mi = 0; mi < 2; ++mi)
#pragma unroll
      for (int ni = 0; ni < 4; ++ni)
        acc[mi][ni] = __builtin_amdgcn_wmma_f32_16x16x32_bf16(
            false, af[mi], false, bf_[ni], (short)0, acc[mi][ni], false, false);
    __syncthreads();
  }

  const int rsub = (lane >> 4) << 3;
  const int ncol = lane & 15;
#pragma unroll
  for (int mi = 0; mi < 2; ++mi)
#pragma unroll
    for (int ni = 0; ni < 4; ++ni)
#pragma unroll
      for (int v = 0; v < 8; ++v) {
        const int m = wm * 32 + mi * 16 + rsub + v;
        y[(size_t)(slot0 + m) * Hc + nbase + wn * 64 + ni * 16 + ncol] = acc[mi][ni][v];
      }
}

// ---------------- combine the token's two expert slots ----------------
__global__ void k_combine(const float* __restrict__ y, const int* __restrict__ slot_of,
                          float* __restrict__ out) {
  const size_t i = (size_t)blockIdx.x * 256 + threadIdx.x;   // over T*H/4
  if (i >= (size_t)Tc * Hc / 4) return;
  const int t = (int)(i / (Hc / 4));
  const int j = (int)(i % (Hc / 4));
  const int sA = slot_of[2 * t];
  const int sB = slot_of[2 * t + 1];
  const float4 a = ((const float4*)(y + (size_t)sA * Hc))[j];
  const float4 b = ((const float4*)(y + (size_t)sB * Hc))[j];
  float4 o;
  o.x = a.x + b.x; o.y = a.y + b.y; o.z = a.z + b.z; o.w = a.w + b.w;
  ((float4*)(out + (size_t)t * Hc))[j] = o;
}

// ---------------- host ----------------
extern "C" void kernel_launch(void* const* d_in, const int* in_sizes, int n_in,
                              void* d_out, int out_size, void* d_ws, size_t ws_size,
                              hipStream_t stream) {
  (void)in_sizes; (void)n_in; (void)out_size; (void)ws_size;
  const float* x  = (const float*)d_in[0];
  const float* gw = (const float*)d_in[1];
  const float* W1 = (const float*)d_in[2];
  const float* W2 = (const float*)d_in[3];
  const float* W3 = (const float*)d_in[4];
  const float* U1 = (const float*)d_in[5];
  const float* V1 = (const float*)d_in[6];
  const float* U2 = (const float*)d_in[7];
  const float* V2 = (const float*)d_in[8];
  const float* U3 = (const float*)d_in[9];
  const float* V3 = (const float*)d_in[10];

  float* out    = (float*)d_out;                 // [T, H] f32
  float* logits = out + (size_t)Tc * Hc;         // [T, E] f32

  char* ws = (char*)d_ws;
  size_t off = 0;
  auto carve = [&](size_t bytes) -> void* {
    void* p = (void*)(ws + off);
    off += (bytes + 255) & ~(size_t)255;
    return p;
  };
  bf16* xb      = (bf16*)carve((size_t)Tc * Hc * 2);
  bf16* W1e     = (bf16*)carve((size_t)Ec * Fc * Hc * 2);
  bf16* W3e     = (bf16*)carve((size_t)Ec * Fc * Hc * 2);
  bf16* W2e     = (bf16*)carve((size_t)Ec * Hc * Fc * 2);
  bf16* hbuf    = (bf16*)carve((size_t)MAXS * Fc * 2);
  float* y      = (float*)carve((size_t)MAXS * Hc * 4);
  int*   sel    = (int*)carve((size_t)Tc * 2 * 4);
  float* wtop   = (float*)carve((size_t)Tc * 2 * 4);
  int*   slotof = (int*)carve((size_t)Tc * 2 * 4);
  int*   tokof  = (int*)carve((size_t)MAXS * 4);
  float* wof    = (float*)carve((size_t)MAXS * 4);
  int*   counts = (int*)carve(64);
  int*   cursor = (int*)carve(64);
  int*   base   = (int*)carve(64);

  k_init<<<(MAXS + 255) / 256, 256, 0, stream>>>(tokof, wof, counts, cursor);
  k_cvt<<<(Tc * Hc / 4 + 255) / 256, 256, 0, stream>>>(x, xb);
  dim3 pg((Fc * Hc + 255) / 256, Ec);
  k_prep<<<pg, 256, 0, stream>>>(W1, U1, V1, W1e, Fc, Hc);
  k_prep<<<pg, 256, 0, stream>>>(W3, U3, V3, W3e, Fc, Hc);
  k_prep<<<pg, 256, 0, stream>>>(W2, U2, V2, W2e, Hc, Fc);
  k_router<<<Tc * 32 / 256, 256, 0, stream>>>(x, gw, logits, sel, wtop, counts);
  k_bases<<<1, 1, 0, stream>>>(counts, base);
  k_assign<<<(Tc + 255) / 256, 256, 0, stream>>>(sel, wtop, base, cursor, tokof, wof, slotof);
  k_gemm1<<<dim3(SLOT_TILES, Fc / 64), 256, 0, stream>>>(xb, W1e, W3e, tokof, wof, base, hbuf);
  k_gemm2<<<dim3(SLOT_TILES, Hc / TILE), 256, 0, stream>>>(hbuf, W2e, tokof, base, y);
  k_combine<<<(Tc * Hc / 4 + 255) / 256, 256, 0, stream>>>(y, slotof, out);
}
